// BellmanFordLayer_38259568673215
// MI455X (gfx1250) — matrix operand order repaired and checked by reading, not compiled, not run on hardware
//
#include <hip/hip_runtime.h>
#include <hip/hip_bf16.h>

#define DIM 256
#define EPW 64          // edges per workgroup (4 M-tiles of 16)

typedef __bf16 bf16_t;
typedef bf16_t v16bf __attribute__((ext_vector_type(16)));
typedef bf16_t v8bf  __attribute__((ext_vector_type(8)));
typedef float  v8f   __attribute__((ext_vector_type(8)));

// ---------------------------------------------------------------------------
// Kernel 1: pack W1,W2 (f32 row-major [K=256][N=256]) into bf16 WMMA-B
// fragment order: flat index i = ((w*8+kt)*16 + nt)*32*16 + lane*16 + e
// holds W[kt*32 + (e<8 ? 8*(lane>>4)+e : 16+8*(lane>>4)+(e-8))][nt*16 + (lane&15)].
// Total 2 * 8*16*32*16 = 131072 bf16 elements (256 KB in d_ws).
// ---------------------------------------------------------------------------
__global__ __launch_bounds__(256) void pack_weights(const float* __restrict__ W1,
                                                    const float* __restrict__ W2,
                                                    bf16_t* __restrict__ wp) {
    int i = blockIdx.x * 256 + threadIdx.x;     // 512 blocks -> 131072 threads
    int w    = i >> 16;
    int rem  = i & 65535;
    int kt   = rem >> 13;
    int rem2 = rem & 8191;
    int nt   = rem2 >> 9;
    int rem3 = rem2 & 511;
    int lane = rem3 >> 4;
    int e    = rem3 & 15;
    int half = lane >> 4;
    int n    = nt * 16 + (lane & 15);
    int k    = kt * 32 + (e < 8 ? half * 8 + e : 16 + half * 8 + (e - 8));
    const float* W = w ? W2 : W1;
    wp[i] = (bf16_t)W[k * DIM + n];
}

// ---------------------------------------------------------------------------
// Kernel 2: out = h  (float4 copy)
// ---------------------------------------------------------------------------
__global__ __launch_bounds__(256) void copy_h(const float4* __restrict__ h,
                                              float4* __restrict__ out, int nvec) {
    int i = blockIdx.x * 256 + threadIdx.x;
    if (i < nvec) out[i] = h[i];
}

// ---------------------------------------------------------------------------
// A-fragment load from row-major bf16 LDS tile (ISA 16-bit A layout:
// lane (r, half) element e: e<8 -> K = 8*half+e ; e>=8 -> K = 16+8*half+(e-8))
// ---------------------------------------------------------------------------
__device__ __forceinline__ v16bf loadA(const bf16_t* __restrict__ base,
                                       int row, int kt, int half) {
    const bf16_t* p = base + row * DIM + kt * 32 + half * 8;
    v8bf lo = *(const v8bf*)p;
    v8bf hi = *(const v8bf*)(p + 16);
    v16bf a;
#pragma unroll
    for (int q = 0; q < 8; ++q) { a[q] = lo[q]; a[8 + q] = hi[q]; }
    return a;
}

// ---------------------------------------------------------------------------
// Kernel 3: per-edge 2-layer MLP with bf16 WMMA + atomic scatter-sum.
// 1 workgroup (128 threads, 4 waves) handles 64 edges (4 M-tiles).
// Each wave owns 4 N-tiles; per (nt,kt) one B fragment feeds 4 WMMAs.
// A fragments gathered into an explicit array per kt so the 8 ds_loads
// can be clause-issued ahead of 4 back-to-back WMMAs.
// ---------------------------------------------------------------------------
__global__ __launch_bounds__(128) void edge_mlp(const float* __restrict__ h,
                                                const int*  __restrict__ eidx,
                                                const float* __restrict__ rel,
                                                const bf16_t* __restrict__ wpack,
                                                const float* __restrict__ b1,
                                                const float* __restrict__ b2,
                                                float* __restrict__ out,
                                                int nE) {
    __shared__ __align__(16) bf16_t lds_x[EPW * DIM];  // x = h[src] + rel (bf16)
    __shared__ __align__(16) bf16_t lds_h[EPW * DIM];  // hid = relu(x@W1+b1)
    __shared__ int s_dst[EPW];

    const int t     = threadIdx.x;
    const int ebase = blockIdx.x * EPW;

    // ---- stage x = h[src] + rel_emb into LDS as bf16, row-major ----------
    {
        int   row   = t >> 1;            // 64 rows, 2 threads per row
        int   c0    = (t & 1) * 128;     // 128 columns per thread
        int   e     = ebase + row;
        bool  valid = e < nE;
        int   ee    = valid ? e : 0;
        int   sidx  = eidx[ee];
        float scale = valid ? 1.0f : 0.0f;
        const float4* hp = (const float4*)(h   + (size_t)sidx * DIM + c0);
        const float4* rp = (const float4*)(rel + (size_t)ee   * DIM + c0);
        bf16_t* xo = &lds_x[row * DIM + c0];
#pragma unroll
        for (int i = 0; i < 16; ++i) {   // 16 chunks of 8 floats
            float4 a0 = hp[2 * i], a1 = hp[2 * i + 1];
            float4 r0 = rp[2 * i], r1 = rp[2 * i + 1];
            float s[8] = {a0.x + r0.x, a0.y + r0.y, a0.z + r0.z, a0.w + r0.w,
                          a1.x + r1.x, a1.y + r1.y, a1.z + r1.z, a1.w + r1.w};
            v8bf v;
#pragma unroll
            for (int q = 0; q < 8; ++q) v[q] = (bf16_t)(s[q] * scale);
            *(v8bf*)(xo + 8 * i) = v;
        }
        if (t < EPW) s_dst[t] = (ebase + t < nE) ? eidx[nE + ebase + t] : -1;
    }
    __syncthreads();

    const int lane = t & 31;
    const int wave = t >> 5;
    const int r    = lane & 15;   // col within N-tile / row within A-tile
    const int half = lane >> 4;

    const v16bf* wb1 = (const v16bf*)wpack;             // GEMM1 B fragments
    const v16bf* wb2 = (const v16bf*)(wpack + 65536);   // GEMM2 B fragments

    // ---- GEMM1: hid = relu(x @ W1 + b1), bf16 into LDS --------------------
#pragma unroll
    for (int j = 0; j < 4; ++j) {
        int   nt   = wave * 4 + j;
        int   n    = nt * 16 + r;
        float bias = b1[n];
        v8f C[4];
#pragma unroll
        for (int mt = 0; mt < 4; ++mt)
#pragma unroll
            for (int g = 0; g < 8; ++g) C[mt][g] = bias;
#pragma unroll
        for (int kt = 0; kt < 8; ++kt) {
            v16bf B = wb1[(size_t)(kt * 16 + nt) * 32 + lane];
            v16bf Am[4];
#pragma unroll
            for (int mt = 0; mt < 4; ++mt)
                Am[mt] = loadA(lds_x, mt * 16 + r, kt, half);
#pragma unroll
            for (int mt = 0; mt < 4; ++mt)
                C[mt] = __builtin_amdgcn_wmma_f32_16x16x32_bf16(
                            false, Am[mt], false, B, (short)0, C[mt], false, false);
        }
#pragma unroll
        for (int mt = 0; mt < 4; ++mt)
#pragma unroll
            for (int g = 0; g < 8; ++g) {
                int m = mt * 16 + half * 8 + g;   // C/D layout: VGPR g -> row
                lds_h[m * DIM + n] = (bf16_t)fmaxf(C[mt][g], 0.0f);
            }
    }
    __syncthreads();

    // ---- GEMM2: msg = hid @ W2 + b2, scatter-add to out[dst] --------------
#pragma unroll
    for (int j = 0; j < 4; ++j) {
        int   nt   = wave * 4 + j;
        int   n    = nt * 16 + r;
        float bias = b2[n];
        v8f C[4];
#pragma unroll
        for (int mt = 0; mt < 4; ++mt)
#pragma unroll
            for (int g = 0; g < 8; ++g) C[mt][g] = bias;
#pragma unroll
        for (int kt = 0; kt < 8; ++kt) {
            v16bf B = wb2[(size_t)(kt * 16 + nt) * 32 + lane];
            v16bf Am[4];
#pragma unroll
            for (int mt = 0; mt < 4; ++mt)
                Am[mt] = loadA(lds_h, mt * 16 + r, kt, half);
#pragma unroll
            for (int mt = 0; mt < 4; ++mt)
                C[mt] = __builtin_amdgcn_wmma_f32_16x16x32_bf16(
                            false, Am[mt], false, B, (short)0, C[mt], false, false);
        }
#pragma unroll
        for (int mt = 0; mt < 4; ++mt)
#pragma unroll
            for (int g = 0; g < 8; ++g) {
                int m = mt * 16 + half * 8 + g;
                int d = s_dst[m];
                if (d >= 0) atomicAdd(out + (size_t)d * DIM + n, C[mt][g]);
            }
    }
}

// ---------------------------------------------------------------------------
extern "C" void kernel_launch(void* const* d_in, const int* in_sizes, int n_in,
                              void* d_out, int out_size, void* d_ws, size_t ws_size,
                              hipStream_t stream) {
    const float* h   = (const float*)d_in[0];
    const int*   ei  = (const int*)d_in[1];   // [2, E] -> src = ei[e], dst = ei[E+e]
    const float* rel = (const float*)d_in[2];
    const float* W1  = (const float*)d_in[3];
    const float* b1  = (const float*)d_in[4];
    const float* W2  = (const float*)d_in[5];
    const float* b2  = (const float*)d_in[6];
    float*  out   = (float*)d_out;
    bf16_t* wpack = (bf16_t*)d_ws;            // 256 KB of bf16 weight fragments
    int nE = in_sizes[1] / 2;

    // 1) pack weights to bf16 WMMA-B fragment order (131072 elements)
    pack_weights<<<512, 256, 0, stream>>>(W1, W2, wpack);

    // 2) out = h
    int nvec = out_size / 4;
    copy_h<<<(nvec + 255) / 256, 256, 0, stream>>>((const float4*)h, (float4*)out, nvec);

    // 3) edge MLP + scatter (64 edges per WG)
    edge_mlp<<<(nE + EPW - 1) / EPW, 128, 0, stream>>>(h, ei, rel, wpack, b1, b2, out, nE);
}
